// MoELayer_62818191671813
// MI455X (gfx1250) — matrix operand order repaired and checked by reading, not compiled, not run on hardware
//
#include <hip/hip_runtime.h>

#define D_IN   4096
#define D_OUTV 4096
#define NE     8
#define T_TOTAL 8192

typedef __attribute__((ext_vector_type(16))) __bf16 v16bf;
typedef __attribute__((ext_vector_type(8)))  __bf16 v8bf;
typedef __attribute__((ext_vector_type(8)))  float  v8f;

typedef unsigned int u32;
typedef __attribute__((ext_vector_type(4))) u32 u32x4;
typedef __attribute__((ext_vector_type(8))) int i32x8;
typedef __attribute__((ext_vector_type(4))) int i32x4;

// ---------------------------------------------------------------------------
// Kernel 1: gating. One 256-thread block per token. logits = x @ gate_w^T + b,
// argmax (softmax is monotone, argmax of logits == argmax of softmax).
// ---------------------------------------------------------------------------
__global__ __launch_bounds__(256) void gate_kernel(const float* __restrict__ x,
                                                   const float* __restrict__ gw,
                                                   const float* __restrict__ gb,
                                                   int* __restrict__ selected) {
  const int token = blockIdx.x;
  const float* xr = x + (size_t)token * D_IN;

  float p[NE];
#pragma unroll
  for (int e = 0; e < NE; ++e) p[e] = 0.0f;

  for (int d = threadIdx.x; d < D_IN; d += 256) {
    const float xv = xr[d];
#pragma unroll
    for (int e = 0; e < NE; ++e) p[e] += xv * gw[e * D_IN + d];
  }

#pragma unroll
  for (int e = 0; e < NE; ++e) {
#pragma unroll
    for (int off = 16; off > 0; off >>= 1) p[e] += __shfl_xor(p[e], off, 32);
  }

  __shared__ float sred[8][NE];
  const int wave = threadIdx.x >> 5;
  const int lane = threadIdx.x & 31;
  if (lane == 0) {
#pragma unroll
    for (int e = 0; e < NE; ++e) sred[wave][e] = p[e];
  }
  __syncthreads();

  if (threadIdx.x == 0) {
    int best = 0;
    float bv = -3.4e38f;
#pragma unroll
    for (int e = 0; e < NE; ++e) {
      float s = gb[e];
#pragma unroll
      for (int w = 0; w < 8; ++w) s += sred[w][e];
      if (s > bv) { bv = s; best = e; }   // first-max tie-break, like jnp.argmax
    }
    selected[token] = best;
  }
}

// ---------------------------------------------------------------------------
// Kernel 2a/2b: per-expert token lists.
// ---------------------------------------------------------------------------
__global__ void zero_counts(int* __restrict__ counts) {
  if (threadIdx.x < NE) counts[threadIdx.x] = 0;
}

__global__ __launch_bounds__(256) void build_lists(const int* __restrict__ selected,
                                                   int* __restrict__ counts,
                                                   int* __restrict__ lists) {
  const int t = blockIdx.x * 256 + threadIdx.x;
  if (t >= T_TOTAL) return;
  const int e = selected[t];
  const int pos = atomicAdd(&counts[e], 1);
  lists[e * T_TOTAL + pos] = t;
}

// ---------------------------------------------------------------------------
// TDM: async-load one 128-row x 32-col fp32 weight tile into LDS.
// D# per ISA 08_async_tensor 8.3/8.4: count=1, type=2 ("image"),
// data_size=4B, tensor 4096x4096, tile 32(x) x 128(y), dim0 stride 4096.
// This toolchain exposes the 6-arg builtin (extra int32x8 group for >2D).
// ---------------------------------------------------------------------------
__device__ __forceinline__ void tdm_load_b_tile(u32 lds_off, const float* gptr) {
  const unsigned long long ga = (unsigned long long)(uintptr_t)gptr;
  u32x4 g0;
  g0[0] = 1u;                                           // count=1, user descriptor
  g0[1] = lds_off;                                      // LDS byte address
  g0[2] = (u32)ga;                                      // global addr [31:0]
  g0[3] = (u32)((ga >> 32) & 0x01FFFFFFu) | (2u << 30); // addr [56:32] | type=2

  i32x8 g1;
  g1[0] = (int)(2u << 16);                              // data_size = 4 bytes
  g1[1] = (int)((4096u & 0xFFFFu) << 16);               // tensor_dim0 lo16 @ [63:48]
  g1[2] = (int)((4096u >> 16) | ((4096u & 0xFFFFu) << 16)); // dim0 hi | dim1 lo
  g1[3] = (int)((4096u >> 16) | (32u << 16));           // dim1 hi | tile_dim0 = 32
  g1[4] = (int)128u;                                    // tile_dim1 = 128, tile_dim2 = 0
  g1[5] = (int)4096u;                                   // tensor_dim0_stride lo32
  g1[6] = 0;                                            // stride0 hi | stride1 lo
  g1[7] = 0;                                            // stride1 hi

  i32x4 z4 = {};
  i32x8 z8 = {};
  __builtin_amdgcn_tensor_load_to_lds(g0, g1, z4, z4, z8, 0); // 2D: groups 2/3 unused
}

// ---------------------------------------------------------------------------
// Kernel 3: grouped GEMM. Block tile: 64 tokens x 128 outputs, K-chunk 32.
// B tile double-buffered through LDS by the Tensor Data Mover (fp32 staging,
// bf16 conversion fused into the fragment build). A tile: global gather ->
// bf16 -> LDS. 8 waves; wave (mw,nw) owns 2x2 WMMA f32_16x16x32_bf16 tiles.
// ---------------------------------------------------------------------------
__global__ __launch_bounds__(256) void moe_gemm(const float* __restrict__ x,
                                                const float* __restrict__ ew,
                                                const float* __restrict__ eb,
                                                const int* __restrict__ counts,
                                                const int* __restrict__ lists,
                                                float* __restrict__ out) {
  const int e = blockIdx.z;
  const int count = counts[e];
  const int m_base = blockIdx.y * 64;
  if (m_base >= count) return;
  const int n_base = blockIdx.x * 128;

  __shared__ __bf16 As[64][32];          // token-rows x K, bf16
  __shared__ float  Bstage[2][128][32];  // TDM staging: out-cols x K, fp32
  __shared__ int    toks[64];

  const int tid = threadIdx.x;

  if (tid < 64) {
    int idx = m_base + tid;
    if (idx >= count) idx = count - 1;   // clamp: never read poisoned list slots
    toks[tid] = lists[e * T_TOTAL + idx];
  }

  const float* btile = ew + ((size_t)e * D_OUTV + (size_t)n_base) * D_IN;
  const u32 ldsB0 = (u32)(uintptr_t)&Bstage[0][0][0];
  const u32 ldsB1 = (u32)(uintptr_t)&Bstage[1][0][0];

  if (tid < 32) tdm_load_b_tile(ldsB0, btile);   // prologue: tile for k0=0

  __syncthreads();                                // toks visible for A gather

  // A staging assignment: 4 threads/row, 8 floats each
  const int a_row = tid >> 2;
  const int a_k   = (tid & 3) * 8;
  const float* xrow = x + (size_t)toks[a_row] * D_IN + a_k;

  const int wave = tid >> 5;
  const int lane = tid & 31;
  const int mw = wave >> 2;              // 0..1
  const int nw = wave & 3;               // 0..3
  const int g   = lane >> 4;             // half-wave
  const int l15 = lane & 15;

  v8f acc[2][2] = {};
  int buf = 0;

  for (int k0 = 0; k0 < D_IN; k0 += 32) {
    // ---- A: global gather (fp32) ----
    const float4 av0 = *(const float4*)(xrow + k0);
    const float4 av1 = *(const float4*)(xrow + k0 + 4);

    if (tid < 32) __builtin_amdgcn_s_wait_tensorcnt(0);  // B tile for this iter landed
    __syncthreads();   // B staged + previous iter's fragment reads complete

    // ---- A: fp32 -> bf16 -> LDS (one 16B store) ----
    {
      v8bf ap;
      ap[0] = (__bf16)av0.x; ap[1] = (__bf16)av0.y; ap[2] = (__bf16)av0.z; ap[3] = (__bf16)av0.w;
      ap[4] = (__bf16)av1.x; ap[5] = (__bf16)av1.y; ap[6] = (__bf16)av1.z; ap[7] = (__bf16)av1.w;
      *(v8bf*)(&As[a_row][a_k]) = ap;
    }

    // ---- kick TDM for next K-chunk into the other buffer ----
    if (tid < 32 && (k0 + 32) < D_IN)
      tdm_load_b_tile(buf ? ldsB0 : ldsB1, btile + k0 + 32);

    __syncthreads();

    // ---- A fragments per ISA 7.12.2 (lane half g: K {8g..8g+7, 16+8g..23+8g}) ----
    v16bf afrag[2];
#pragma unroll
    for (int mi = 0; mi < 2; ++mi) {
      const int row = (mw * 2 + mi) * 16 + l15;
      const v8bf lo = *(const v8bf*)(&As[row][g * 8]);
      const v8bf hi = *(const v8bf*)(&As[row][g * 8 + 16]);
      v16bf a;
#pragma unroll
      for (int i = 0; i < 8; ++i) { a[i] = lo[i]; a[i + 8] = hi[i]; }
      afrag[mi] = a;
    }

    // ---- B fragments: fp32 from TDM staging, cvt fused (lane half g: K {8g..8g+7, 16+8g..23+8g}) ----
    v16bf bfrag[2];
#pragma unroll
    for (int ni = 0; ni < 2; ++ni) {
      const int col = (nw * 2 + ni) * 16 + l15;
      const float* bp = &Bstage[buf][col][g * 8];
      const float4 f0 = *(const float4*)(bp + 0);
      const float4 f1 = *(const float4*)(bp + 4);
      const float4 f2 = *(const float4*)(bp + 16);
      const float4 f3 = *(const float4*)(bp + 20);
      v16bf b;
      b[0]  = (__bf16)f0.x; b[1]  = (__bf16)f0.y; b[2]  = (__bf16)f0.z; b[3]  = (__bf16)f0.w;
      b[4]  = (__bf16)f1.x; b[5]  = (__bf16)f1.y; b[6]  = (__bf16)f1.z; b[7]  = (__bf16)f1.w;
      b[8]  = (__bf16)f2.x; b[9]  = (__bf16)f2.y; b[10] = (__bf16)f2.z; b[11] = (__bf16)f2.w;
      b[12] = (__bf16)f3.x; b[13] = (__bf16)f3.y; b[14] = (__bf16)f3.z; b[15] = (__bf16)f3.w;
      bfrag[ni] = b;
    }

    // ---- 4 WMMAs ----
#pragma unroll
    for (int mi = 0; mi < 2; ++mi)
#pragma unroll
      for (int ni = 0; ni < 2; ++ni)
        acc[mi][ni] = __builtin_amdgcn_wmma_f32_16x16x32_bf16(
            false, afrag[mi], false, bfrag[ni], (short)0, acc[mi][ni], false, false);

    buf ^= 1;
  }

  // ---- epilogue: vector token fetch, fast path for full tiles ----
  const bool full = (m_base + 64) <= count;
#pragma unroll
  for (int mi = 0; mi < 2; ++mi) {
    const int rbase = (mw * 2 + mi) * 16 + g * 8;       // C layout: VGPR v -> row rbase+v
    const int4 ta = *(const int4*)(&toks[rbase]);
    const int4 tb = *(const int4*)(&toks[rbase + 4]);
    const int tok[8] = { ta.x, ta.y, ta.z, ta.w, tb.x, tb.y, tb.z, tb.w };
#pragma unroll
    for (int ni = 0; ni < 2; ++ni) {
      const int ncol = n_base + (nw * 2 + ni) * 16 + l15;
      const float bias = eb[e * D_OUTV + ncol];
      if (full) {
#pragma unroll
        for (int v = 0; v < 8; ++v)
          out[(size_t)tok[v] * D_OUTV + ncol] = acc[mi][ni][v] + bias;
      } else {
#pragma unroll
        for (int v = 0; v < 8; ++v)
          if (m_base + rbase + v < count)
            out[(size_t)tok[v] * D_OUTV + ncol] = acc[mi][ni][v] + bias;
      }
    }
  }
}

// ---------------------------------------------------------------------------
// Launch. ws layout: selected[8192] | counts[8] (padded to 64) | lists[8*8192]
// = ~295 KB of d_ws.
// ---------------------------------------------------------------------------
extern "C" void kernel_launch(void* const* d_in, const int* in_sizes, int n_in,
                              void* d_out, int out_size, void* d_ws, size_t ws_size,
                              hipStream_t stream) {
  (void)in_sizes; (void)n_in; (void)out_size; (void)ws_size;
  const float* x  = (const float*)d_in[0];
  const float* gw = (const float*)d_in[1];
  const float* gb = (const float*)d_in[2];
  const float* ew = (const float*)d_in[3];
  const float* eb = (const float*)d_in[4];
  float* out = (float*)d_out;

  int* selected = (int*)d_ws;
  int* counts   = selected + T_TOTAL;
  int* lists    = counts + 64;

  gate_kernel<<<T_TOTAL, 256, 0, stream>>>(x, gw, gb, selected);
  zero_counts<<<1, 8, 0, stream>>>(counts);
  build_lists<<<T_TOTAL / 256, 256, 0, stream>>>(selected, counts, lists);

  dim3 grid(D_OUTV / 128, T_TOTAL / 64, NE);  // expert slowest -> L2 locality
  moe_gemm<<<grid, 256, 0, stream>>>(x, ew, eb, counts, lists, out);
}
